// MambaBlock_8358006358375
// MI455X (gfx1250) — matrix operand order repaired and checked by reading, not compiled, not run on hardware
//
#include <hip/hip_runtime.h>
#include <hip/hip_bf16.h>
#include <math.h>

// ---- problem dimensions (from the reference) ----
#define B_   2
#define L_   2048
#define DM   1024
#define DS   64
#define DI   2048          // DM * EXP
#define DC   4
#define NE   (2*DS + 1)    // 129 rows of W_x
#define ML   (B_ * L_)     // 4096 GEMM rows
#define CL   64            // scan chunk length
#define NCH  (L_ / CL)     // 32 chunks

typedef __attribute__((ext_vector_type(16))) __bf16 v16bf;
typedef __attribute__((ext_vector_type(8)))  float  v8f;

__device__ __forceinline__ float sigmoidf_(float x) { return 1.0f / (1.0f + __expf(-x)); }

// ---------------------------------------------------------------------------
// f32 -> bf16 weight conversion (done once per weight matrix)
// ---------------------------------------------------------------------------
__global__ void f32_to_bf16_kernel(const float* __restrict__ in, __bf16* __restrict__ out, int n) {
    int i = blockIdx.x * blockDim.x + threadIdx.x;
    if (i < n) out[i] = (__bf16)in[i];
}

// ---------------------------------------------------------------------------
// C[M,N] = A[M,K] (f32, row-major) x Bw[N,K]^T (bf16, row-major)
// via V_WMMA_F32_16X16X32_BF16.
// Block = 256 threads = 8 waves (2M x 4N), block tile 64x256.
// Wave tile = 32x64: 2 A-fragments x 4 B-fragments -> 8 WMMAs per K-step,
// so each (expensive) f32 A-fragment load+convert feeds 4 WMMAs.
// Fragment layouts used here (A and B share one k order):
//   A (16-bit, 16x32): lane&15 = M, half = lane>>4, elem e -> K = half*16 + e
//   B: lane&15 = N, elem e -> K = half*16 + e  (16 contiguous bf16 = one 32B load)
//   C/D: elem r -> M = r + 8*half, N = lane&15
// ---------------------------------------------------------------------------
__global__ __launch_bounds__(256)
void gemm_wmma_kernel(const float* __restrict__ A, const __bf16* __restrict__ Bw,
                      float* __restrict__ C, int M, int N, int K,
                      int lda, int ldb, int ldc)
{
    const int lane = threadIdx.x & 31;
    const int wave = threadIdx.x >> 5;            // 0..7
    const int half = lane >> 4;
    const int lm   = lane & 15;
    const int row0 = blockIdx.y * 64  + (wave >> 2) * 32;   // 2 wave-rows
    const int col0 = blockIdx.x * 256 + (wave & 3) * 64;    // 4 wave-cols

    v8f acc[2][4];
    #pragma unroll
    for (int i = 0; i < 2; ++i)
        #pragma unroll
        for (int j = 0; j < 4; ++j) acc[i][j] = (v8f){};

    const float* Arow[2];
    #pragma unroll
    for (int i = 0; i < 2; ++i)
        Arow[i] = A + (size_t)(row0 + i * 16 + lm) * lda;

    const __bf16* Brow[4];
    #pragma unroll
    for (int j = 0; j < 4; ++j) {
        int nb = col0 + j * 16 + lm;
        if (nb >= N) nb = N - 1;                  // clamp for ragged N
        Brow[j] = Bw + (size_t)nb * ldb;
    }

    for (int k0 = 0; k0 < K; k0 += 32) {
        // prefetch next K-slab of the f32 activation stream (global_prefetch_b8)
        if (k0 + 32 < K) {
            __builtin_prefetch(Arow[0] + k0 + 32, 0, 1);
            __builtin_prefetch(Arow[1] + k0 + 32, 0, 1);
        }

        // A fragments: four 16B vector loads each, convert f32 -> bf16 packed
        v16bf a[2];
        #pragma unroll
        for (int i = 0; i < 2; ++i) {
            const float4* pa = (const float4*)(Arow[i] + k0 + half * 16);
            float4 a0 = pa[0];   // K = base + 0..3
            float4 a1 = pa[1];   // K = base + 4..7
            float4 a2 = pa[2];   // K = base + 8..11
            float4 a3 = pa[3];   // K = base + 12..15
            a[i][0]=(__bf16)a0.x; a[i][1]=(__bf16)a0.y; a[i][2]=(__bf16)a0.z; a[i][3]=(__bf16)a0.w;
            a[i][4]=(__bf16)a1.x; a[i][5]=(__bf16)a1.y; a[i][6]=(__bf16)a1.z; a[i][7]=(__bf16)a1.w;
            a[i][8]=(__bf16)a2.x; a[i][9]=(__bf16)a2.y; a[i][10]=(__bf16)a2.z; a[i][11]=(__bf16)a2.w;
            a[i][12]=(__bf16)a3.x; a[i][13]=(__bf16)a3.y; a[i][14]=(__bf16)a3.z; a[i][15]=(__bf16)a3.w;
        }

        // B fragments: 16 contiguous bf16 (32 bytes, 32B-aligned) each
        #pragma unroll
        for (int j = 0; j < 4; ++j) {
            v16bf bfv = *(const v16bf*)(Brow[j] + k0 + half * 16);
            acc[0][j] = __builtin_amdgcn_wmma_f32_16x16x32_bf16(
                            false, a[0], false, bfv, (short)0, acc[0][j], false, false);
            acc[1][j] = __builtin_amdgcn_wmma_f32_16x16x32_bf16(
                            false, a[1], false, bfv, (short)0, acc[1][j], false, false);
        }
    }

    #pragma unroll
    for (int i = 0; i < 2; ++i) {
        #pragma unroll
        for (int j = 0; j < 4; ++j) {
            const int nc = col0 + j * 16 + lm;
            if (nc < N) {
                #pragma unroll
                for (int r = 0; r < 8; ++r) {
                    int mr = row0 + i * 16 + half * 8 + r;
                    C[(size_t)mr * ldc + nc] = acc[i][j][r];
                }
            }
        }
    }
}

// ---------------------------------------------------------------------------
// Depthwise causal conv (k=4) + bias + SiLU on x_in (= xz[..., :DI])
// ---------------------------------------------------------------------------
__global__ void conv_silu_kernel(const float* __restrict__ xz, const float* __restrict__ cw,
                                 const float* __restrict__ cb, float* __restrict__ xc)
{
    int i = blockIdx.x * blockDim.x + threadIdx.x;     // over B_*L_*DI
    if (i >= B_ * L_ * DI) return;
    int d = i % DI;
    int l = (i / DI) % L_;
    int b = i / (DI * L_);
    float acc = cb[d];
    #pragma unroll
    for (int j = 0; j < DC; ++j) {
        int ls = l - (DC - 1) + j;
        if (ls >= 0)
            acc += cw[d * DC + j] * xz[((size_t)(b * L_ + ls)) * (2 * DI) + d];
    }
    xc[i] = acc * sigmoidf_(acc);
}

// ---------------------------------------------------------------------------
// Scan pass A: chunk-local scans with h_start = 0.
// Writes y_local, per-chunk end state h_end, and cumulative A_bar products g.
// grid = (DI/256, NCH, B_), block = 256 threads over d.
// ---------------------------------------------------------------------------
__global__ __launch_bounds__(256)
void scan_local_kernel(const float* __restrict__ xc, const float* __restrict__ ssm,
                       const float* __restrict__ A_log, float* __restrict__ y,
                       float* __restrict__ hend, float* __restrict__ g)
{
    __shared__ float Bs[DS], Cs[DS], Abar[DS], Aneg[DS];
    __shared__ float dt_s;
    const int tid = threadIdx.x;
    const int d   = blockIdx.x * 256 + tid;
    const int c   = blockIdx.y;
    const int b   = blockIdx.z;

    if (tid < DS) Aneg[tid] = -__expf(A_log[tid]);
    float h[DS];
    #pragma unroll
    for (int s = 0; s < DS; ++s) h[s] = 0.f;
    float grun = 1.f;
    __syncthreads();

    for (int tt = 0; tt < CL; ++tt) {
        const int t = c * CL + tt;
        const float* row = ssm + (size_t)(b * L_ + t) * NE;
        if (tid < DS)               Bs[tid]        = row[tid];
        else if (tid < 2 * DS)      Cs[tid - DS]   = row[tid];
        else if (tid == 2 * DS) {
            float v = row[2 * DS];
            dt_s = (v > 20.f) ? v : log1pf(__expf(v));   // softplus
        }
        __syncthreads();
        if (tid < DS) Abar[tid] = __expf(Aneg[tid] * dt_s);
        __syncthreads();

        const float xdt = xc[(size_t)(b * L_ + t) * DI + d] * dt_s;
        float acc = 0.f;
        #pragma unroll
        for (int s = 0; s < DS; ++s) {
            h[s] = Abar[s] * h[s] + xdt * Bs[s];
            acc += h[s] * Cs[s];
        }
        y[(size_t)(b * L_ + t) * DI + d] = acc;

        if (tid < DS) {
            grun *= Abar[tid];
            if (blockIdx.x == 0) g[((size_t)(b * L_) + t) * DS + tid] = grun;
        }
        __syncthreads();
    }

    float* hp = hend + (((size_t)(b * NCH + c) * DI + d) * DS);
    #pragma unroll
    for (int s = 0; s < DS; ++s) hp[s] = h[s];
}

// ---------------------------------------------------------------------------
// Scan pass B: sequential carry across the 32 chunks; converts hbuf from
// per-chunk end-states to per-chunk start-states in place (thread-private).
// One thread per (b, d, s) -> 262144 independent threads, 32-step loop.
// ---------------------------------------------------------------------------
__global__ void scan_carry_kernel(const float* __restrict__ g, float* __restrict__ hbuf)
{
    int i = blockIdx.x * blockDim.x + threadIdx.x;     // over B_*DI*DS
    if (i >= B_ * DI * DS) return;
    int s = i % DS;
    int d = (i / DS) % DI;
    int b = i / (DS * DI);
    float hs = 0.f;
    for (int c = 0; c < NCH; ++c) {
        size_t idx = ((size_t)(b * NCH + c) * DI + d) * DS + s;
        float he = hbuf[idx];
        hbuf[idx] = hs;                                // h_start for chunk c
        float G = g[((size_t)b * L_ + (c * CL + CL - 1)) * DS + s];
        hs = G * hs + he;
    }
}

// ---------------------------------------------------------------------------
// Scan pass C: add carried-state contribution  sum_s C_t[s]*g_t[s]*h_start[s],
// fused with the D-residual and SiLU(z) gating.  Same geometry as pass A.
// ---------------------------------------------------------------------------
__global__ __launch_bounds__(256)
void scan_correct_kernel(const float* __restrict__ ssm, const float* __restrict__ g,
                         const float* __restrict__ hstart, const float* __restrict__ xc,
                         const float* __restrict__ xz, const float* __restrict__ Dv,
                         float* __restrict__ y)
{
    __shared__ float Cs[DS], gs[DS];
    const int tid = threadIdx.x;
    const int d   = blockIdx.x * 256 + tid;
    const int c   = blockIdx.y;
    const int b   = blockIdx.z;

    float hs[DS];
    const float* hp = hstart + (((size_t)(b * NCH + c) * DI + d) * DS);
    #pragma unroll
    for (int s = 0; s < DS; ++s) hs[s] = hp[s];
    const float Dd = Dv[d];

    for (int tt = 0; tt < CL; ++tt) {
        const int t = c * CL + tt;
        const float* row = ssm + (size_t)(b * L_ + t) * NE;
        if (tid < DS) {
            Cs[tid] = row[DS + tid];
            gs[tid] = g[((size_t)(b * L_) + t) * DS + tid];
        }
        __syncthreads();
        float corr = 0.f;
        #pragma unroll
        for (int s = 0; s < DS; ++s) corr += Cs[s] * gs[s] * hs[s];

        size_t i = (size_t)(b * L_ + t) * DI + d;
        float v = y[i] + corr + xc[i] * Dd;
        float z = xz[(size_t)(b * L_ + t) * (2 * DI) + DI + d];
        y[i] = v * (z * sigmoidf_(z));
        __syncthreads();
    }
}

// ---------------------------------------------------------------------------
extern "C" void kernel_launch(void* const* d_in, const int* in_sizes, int n_in,
                              void* d_out, int out_size, void* d_ws, size_t ws_size,
                              hipStream_t stream)
{
    (void)in_sizes; (void)n_in; (void)out_size; (void)ws_size;

    const float* x     = (const float*)d_in[0];   // (B,L,DM)
    const float* W_in  = (const float*)d_in[1];   // (2*DI, DM)
    const float* convw = (const float*)d_in[2];   // (DI,1,DC)
    const float* convb = (const float*)d_in[3];   // (DI,)
    const float* W_x   = (const float*)d_in[4];   // (NE, DI)
    const float* A_log = (const float*)d_in[5];   // (DS,)
    const float* Dv    = (const float*)d_in[6];   // (DI,)
    const float* W_out = (const float*)d_in[7];   // (DM, DI)
    float* out = (float*)d_out;                   // (B,L,DM) f32

    char* ws = (char*)d_ws;
    size_t off = 0;
    auto take = [&](size_t bytes) -> char* {
        char* p = ws + off;
        off += (bytes + 255) & ~(size_t)255;
        return p;
    };
    float*  xz    = (float*)  take((size_t)ML * 2 * DI * sizeof(float));        // 64 MB
    float*  xc    = (float*)  take((size_t)ML * DI * sizeof(float));            // 32 MB
    float*  ssm   = (float*)  take((size_t)ML * NE * sizeof(float));            //  2 MB
    float*  y     = (float*)  take((size_t)ML * DI * sizeof(float));            // 32 MB
    float*  hbuf  = (float*)  take((size_t)B_ * NCH * DI * DS * sizeof(float)); // 32 MB
    float*  g     = (float*)  take((size_t)B_ * L_ * DS * sizeof(float));       //  1 MB
    __bf16* Winb  = (__bf16*) take((size_t)2 * DI * DM * sizeof(__bf16));       //  8 MB
    __bf16* Wxb   = (__bf16*) take((size_t)NE * DI * sizeof(__bf16));           // ~0.5 MB
    __bf16* Woutb = (__bf16*) take((size_t)DM * DI * sizeof(__bf16));           //  4 MB

    // weight conversions f32 -> bf16
    {
        int n1 = 2 * DI * DM, n2 = NE * DI, n3 = DM * DI;
        f32_to_bf16_kernel<<<(n1 + 255) / 256, 256, 0, stream>>>(W_in,  Winb,  n1);
        f32_to_bf16_kernel<<<(n2 + 255) / 256, 256, 0, stream>>>(W_x,   Wxb,   n2);
        f32_to_bf16_kernel<<<(n3 + 255) / 256, 256, 0, stream>>>(W_out, Woutb, n3);
    }

    // 1) xz = x @ W_in^T : M=4096, N=4096, K=1024   (block tile 64x256)
    gemm_wmma_kernel<<<dim3((2 * DI) / 256, ML / 64), 256, 0, stream>>>(
        x, Winb, xz, ML, 2 * DI, DM, DM, DM, 2 * DI);

    // 2) depthwise causal conv + SiLU
    conv_silu_kernel<<<(ML * DI + 255) / 256, 256, 0, stream>>>(xz, convw, convb, xc);

    // 3) ssm = xc @ W_x^T : M=4096, N=129, K=2048
    gemm_wmma_kernel<<<dim3((NE + 255) / 256, ML / 64), 256, 0, stream>>>(
        xc, Wxb, ssm, ML, NE, DI, DI, DI, NE);

    // 4) chunked selective scan: local -> carry -> correct(+residual+gate)
    scan_local_kernel<<<dim3(DI / 256, NCH, B_), 256, 0, stream>>>(
        xc, ssm, A_log, y, hbuf, g);
    scan_carry_kernel<<<(B_ * DI * DS + 255) / 256, 256, 0, stream>>>(g, hbuf);
    scan_correct_kernel<<<dim3(DI / 256, NCH, B_), 256, 0, stream>>>(
        ssm, g, hbuf, xc, xz, Dv, y);

    // 5) out = y @ W_out^T : M=4096, N=1024, K=2048
    gemm_wmma_kernel<<<dim3(DM / 256, ML / 64), 256, 0, stream>>>(
        y, Woutb, out, ML, DM, DI, DI, DI, DM);
}
